// FlashAttentionSim_65687229825379
// MI455X (gfx1250) — matrix operand order repaired and checked by reading, not compiled, not run on hardware
//
#include <hip/hip_runtime.h>

// ---------------------------------------------------------------------------
// Types for CDNA5 WMMA (wave32): v_wmma_f32_16x16x32_bf16
// ---------------------------------------------------------------------------
typedef __bf16 bf16_t;
typedef bf16_t v16bf __attribute__((ext_vector_type(16)));
typedef bf16_t v8bf  __attribute__((ext_vector_type(8)));
typedef bf16_t v4bf  __attribute__((ext_vector_type(4)));
typedef float  v8f   __attribute__((ext_vector_type(8)));
typedef unsigned int u32x4 __attribute__((ext_vector_type(4)));
typedef int          i32x4 __attribute__((ext_vector_type(4)));
typedef int          i32x8 __attribute__((ext_vector_type(8)));

#define D_MODEL 1024
#define N_HEADS 16
#define D_HEAD  64
#define SEQ_T   2048
#define BATCH   2
#define M_ROWS  (BATCH * SEQ_T)   // 4096

__device__ __forceinline__ bf16_t f2bf(float f) {
  unsigned u = __builtin_bit_cast(unsigned, f);
  unsigned r = (u + 0x7FFFu + ((u >> 16) & 1u)) >> 16;
  unsigned short s = (unsigned short)r;
  return __builtin_bit_cast(bf16_t, s);
}

// Build a 16-element bf16 fragment from two contiguous 16-byte chunks.
__device__ __forceinline__ v16bf ld_frag(const bf16_t* p0, const bf16_t* p1) {
  v8bf lo = *(const v8bf*)p0;
  v8bf hi = *(const v8bf*)p1;
  return __builtin_shufflevector(lo, hi, 0,1,2,3,4,5,6,7,8,9,10,11,12,13,14,15);
}

// ds_swizzle xor-butterfly within 16-lane half groups (group-of-32 mode).
template<int PAT>
__device__ __forceinline__ float swz(float x) {
  int i = __builtin_amdgcn_ds_swizzle(__builtin_bit_cast(int, x), 0x7C00 | PAT);
  return __builtin_bit_cast(float, i);
}
__device__ __forceinline__ float rowmax16(float v) {
  v = fmaxf(v, swz<1>(v)); v = fmaxf(v, swz<2>(v));
  v = fmaxf(v, swz<4>(v)); v = fmaxf(v, swz<8>(v));
  return v;
}
__device__ __forceinline__ float rowsum16(float v) {
  v += swz<1>(v); v += swz<2>(v); v += swz<4>(v); v += swz<8>(v);
  return v;
}

// ---------------------------------------------------------------------------
// TDM: async 2-D tile load (bf16) global -> LDS with in-flight row padding.
// D# per CDNA5 ISA. data_size=2B. Rows of tile_d0=32
// elements (64 B); pad_interval=3 (every 16 DWORDs) + pad_amount=3 (4 DWORDs)
// -> LDS row stride 80 B, matching bf16_t[...][40] tiles.
// 6-arg builtin (this toolchain): (g0, g1, g2, g3, g4, cpol).
// ---------------------------------------------------------------------------
__device__ __forceinline__ void tdm_load_tile(
    unsigned lds_addr, const bf16_t* gsrc,
    unsigned tile_d0, unsigned tile_d1, unsigned stride_elems) {
  unsigned long long ga = (unsigned long long)(size_t)gsrc;
  u32x4 g0;
  g0[0] = 1u;                                            // count=1, user desc
  g0[1] = lds_addr;                                      // lds_addr (bytes)
  g0[2] = (unsigned)ga;                                  // global_addr[31:0]
  g0[3] = ((unsigned)(ga >> 32) & 0x01FFFFFFu) | (2u << 30);  // [56:32] | type=2
  i32x8 g1;
  g1[0] = (int)((1u << 16)      // data_size = 1 -> 2 bytes
              | (1u << 20)      // pad_enable
              | (3u << 22)      // pad_interval: 16 DWORDs (64 B)
              | (3u << 25));    // pad_amount:   4 DWORDs (16 B)
  unsigned td0 = stride_elems;  // tensor_dim0 (in-bounds tiles only)
  unsigned td1 = 1u << 22;      // tensor_dim1 (large, in-bounds tiles only)
  g1[1] = (int)((td0 & 0xFFFFu) << 16);
  g1[2] = (int)(((td0 >> 16) & 0xFFFFu) | ((td1 & 0xFFFFu) << 16));
  g1[3] = (int)(((td1 >> 16) & 0xFFFFu) | ((tile_d0 & 0xFFFFu) << 16));
  g1[4] = (int)(tile_d1 & 0xFFFFu);                      // tile_dim1; tile_dim2=0
  g1[5] = (int)stride_elems;                             // tensor_dim0_stride[31:0]
  g1[6] = 0;
  g1[7] = 0;
  i32x4 zz4 = (i32x4){0, 0, 0, 0};                       // 2-D: groups 2/3 unused
  i32x8 zz8 = (i32x8){0, 0, 0, 0, 0, 0, 0, 0};
  __builtin_amdgcn_tensor_load_to_lds(g0, g1, zz4, zz4, zz8, 0);
}

// ---------------------------------------------------------------------------
// f32 -> bf16 bulk convert (one-time; feeds the TDM which copies raw bytes).
// ---------------------------------------------------------------------------
__global__ __launch_bounds__(256) void cvt_bf16(
    const float* __restrict__ s, bf16_t* __restrict__ d, int n) {
  int i = (blockIdx.x * 256 + threadIdx.x) * 4;
  if (i < n) {
    float4 v = *(const float4*)(s + i);
    v4bf o = { f2bf(v.x), f2bf(v.y), f2bf(v.z), f2bf(v.w) };
    *(v4bf*)(d + i) = o;
  }
}

// ---------------------------------------------------------------------------
// GEMM: Y = A @ W^T. A: (4096x1024) bf16 row-major, W: (1024x1024) bf16
// row-major. Block: 256 thr (8 waves), 128x128 tile; wave = 32x64 (2x4 WMMA).
// Tiles staged by TDM with double buffering (prefetch k+1 under WMMA on k).
// mode 0: f32 row-major | mode 1: bf16 (B,H,T,64) | mode 2: bf16 (B,H,64,T)
// ---------------------------------------------------------------------------
__global__ __launch_bounds__(256) void gemm_wmma(
    const bf16_t* __restrict__ A, const bf16_t* __restrict__ W,
    void* __restrict__ Out, int mode) {
  __shared__ __align__(16) bf16_t Xs[2][128][40];   // A tile, [m][k]
  __shared__ __align__(16) bf16_t Wsh[2][128][40];  // W tile, [n][k]

  const int tid  = threadIdx.x;
  const int w    = tid >> 5;
  const int lane = tid & 31;
  const int half = lane >> 4;
  const int l15  = lane & 15;
  const int wrow = (w & 3) * 32;
  const int wcol = (w >> 2) * 64;
  const int m0   = blockIdx.x * 128;
  const int n0   = blockIdx.y * 128;

  v8f acc[2][4];
#pragma unroll
  for (int i = 0; i < 2; i++)
#pragma unroll
    for (int j = 0; j < 4; j++) acc[i][j] = (v8f){};

  const int NIT = D_MODEL / 32;
  if (w == 0) {  // wave 0 drives the TDM (tensor ops ignore EXEC; wave-level op)
    tdm_load_tile((unsigned)(size_t)&Xs[0][0][0],  A + (size_t)m0 * D_MODEL, 32, 128, D_MODEL);
    tdm_load_tile((unsigned)(size_t)&Wsh[0][0][0], W + (size_t)n0 * D_MODEL, 32, 128, D_MODEL);
  }

  for (int it = 0; it < NIT; it++) {
    if (w == 0) {
      if (it + 1 < NIT) {
        int k1 = (it + 1) * 32;
        int nb = (it + 1) & 1;
        tdm_load_tile((unsigned)(size_t)&Xs[nb][0][0],
                      A + (size_t)m0 * D_MODEL + k1, 32, 128, D_MODEL);
        tdm_load_tile((unsigned)(size_t)&Wsh[nb][0][0],
                      W + (size_t)n0 * D_MODEL + k1, 32, 128, D_MODEL);
        __builtin_amdgcn_s_wait_tensorcnt(2);   // tiles for iter `it` complete
      } else {
        __builtin_amdgcn_s_wait_tensorcnt(0);
      }
    }
    __syncthreads();

    const int buf = it & 1;
    v16bf a[2], b[4];
#pragma unroll
    for (int i = 0; i < 2; i++) {
      const bf16_t* xr = &Xs[buf][wrow + i * 16 + l15][0];
      a[i] = ld_frag(xr + 16 * half, xr + 16 * half + 8);
    }
#pragma unroll
    for (int j = 0; j < 4; j++) {
      const bf16_t* wr = &Wsh[buf][wcol + j * 16 + l15][0];
      b[j] = ld_frag(wr + 16 * half, wr + 16 * half + 8);
    }
#pragma unroll
    for (int i = 0; i < 2; i++)
#pragma unroll
      for (int j = 0; j < 4; j++)
        acc[i][j] = __builtin_amdgcn_wmma_f32_16x16x32_bf16(
            false, a[i], false, b[j], (short)0, acc[i][j], false, false);
    __syncthreads();   // safe to overwrite buf two stages later
  }

  // ---- store ----
#pragma unroll
  for (int i = 0; i < 2; i++) {
#pragma unroll
    for (int j = 0; j < 4; j++) {
      int gm0 = m0 + wrow + i * 16 + 8 * half;
      int gn  = n0 + wcol + j * 16 + l15;
#pragma unroll
      for (int v = 0; v < 8; v++) {
        int gm = gm0 + v;
        float val = acc[i][j][v];
        if (mode == 0) {
          ((float*)Out)[(size_t)gm * D_MODEL + gn] = val;
        } else {
          int bb = gm >> 11, t = gm & (SEQ_T - 1);
          int h  = gn >> 6,  dd = gn & 63;
          bf16_t bv = f2bf(val);
          if (mode == 1)
            ((bf16_t*)Out)[(((size_t)(bb * N_HEADS + h) * SEQ_T + t) << 6) + dd] = bv;
          else
            ((bf16_t*)Out)[(((size_t)(bb * N_HEADS + h) * D_HEAD + dd) << 11) + t] = bv;
        }
      }
    }
  }
}

// ---------------------------------------------------------------------------
// Flash attention (causal). Q,K: bf16 (B,H,T,64); Vt: bf16 (B,H,64,T).
// 8 waves/block; each wave owns a 16-row q tile, streams 32 keys/step.
// Online softmax in registers (ds_swizzle row reductions); P converted
// C-layout -> A-layout through a per-wave LDS slab. O: bf16 (4096x1024).
// ---------------------------------------------------------------------------
__global__ __launch_bounds__(256) void flash_attn(
    const bf16_t* __restrict__ Q, const bf16_t* __restrict__ K,
    const bf16_t* __restrict__ Vt, bf16_t* __restrict__ O) {
  __shared__ __align__(16) bf16_t Pl[8][16][40];

  const int tid  = threadIdx.x;
  const int w    = tid >> 5;
  const int lane = tid & 31;
  const int half = lane >> 4;
  const int l15  = lane & 15;
  const int bh   = blockIdx.x;
  const int q0   = (blockIdx.y * 8 + w) * 16;

  const bf16_t* Qb = Q  + (size_t)bh * SEQ_T * D_HEAD;
  const bf16_t* Kb = K  + (size_t)bh * SEQ_T * D_HEAD;
  const bf16_t* Vb = Vt + (size_t)bh * D_HEAD * SEQ_T;

  v16bf aq[2];
  {
    const bf16_t* qr = Qb + (size_t)(q0 + l15) * D_HEAD;
    aq[0] = ld_frag(qr +      16 * half, qr +      16 * half + 8);
    aq[1] = ld_frag(qr + 32 + 16 * half, qr + 32 + 16 * half + 8);
  }

  v8f acc[4];
#pragma unroll
  for (int c = 0; c < 4; c++) acc[c] = (v8f){};
  float mrow[8], lrow[8];
#pragma unroll
  for (int v = 0; v < 8; v++) { mrow[v] = -1e30f; lrow[v] = 0.0f; }

  for (int kv0 = 0; kv0 < q0 + 16; kv0 += 32) {
    v8f Sj[2];
#pragma unroll
    for (int j = 0; j < 2; j++) {
      const bf16_t* kr = Kb + (size_t)(kv0 + j * 16 + l15) * D_HEAD;
      v16bf bk0 = ld_frag(kr +      16 * half, kr +      16 * half + 8);
      v16bf bk1 = ld_frag(kr + 32 + 16 * half, kr + 32 + 16 * half + 8);
      v8f z = (v8f){};
      z = __builtin_amdgcn_wmma_f32_16x16x32_bf16(false, aq[0], false, bk0,
                                                  (short)0, z, false, false);
      z = __builtin_amdgcn_wmma_f32_16x16x32_bf16(false, aq[1], false, bk1,
                                                  (short)0, z, false, false);
      Sj[j] = z;
    }

#pragma unroll
    for (int v = 0; v < 8; v++) {
      int qpos = q0 + v + 8 * half;
#pragma unroll
      for (int j = 0; j < 2; j++) {
        int kvp = kv0 + j * 16 + l15;
        float s = Sj[j][v] * 0.125f;           // 1/sqrt(64)
        Sj[j][v] = (kvp > qpos) ? -1e30f : s;
      }
      float pm = rowmax16(fmaxf(Sj[0][v], Sj[1][v]));
      float mnew = fmaxf(mrow[v], pm);
      float corr = __expf(mrow[v] - mnew);
      mrow[v] = mnew;
      float p0 = __expf(Sj[0][v] - mnew);
      float p1 = __expf(Sj[1][v] - mnew);
      Sj[0][v] = p0; Sj[1][v] = p1;
      lrow[v] = corr * lrow[v] + rowsum16(p0 + p1);
#pragma unroll
      for (int c = 0; c < 4; c++) acc[c][v] *= corr;
    }

#pragma unroll
    for (int v = 0; v < 8; v++) {
#pragma unroll
      for (int j = 0; j < 2; j++)
        Pl[w][v + 8 * half][j * 16 + l15] = f2bf(Sj[j][v]);
    }
    const bf16_t* pr = &Pl[w][l15][0];
    v16bf ap = ld_frag(pr + 16 * half, pr + 16 * half + 8);

#pragma unroll
    for (int c = 0; c < 4; c++) {
      const bf16_t* vr = Vb + (size_t)(c * 16 + l15) * SEQ_T + kv0 + 16 * half;
      v16bf bv = ld_frag(vr, vr + 8);
      acc[c] = __builtin_amdgcn_wmma_f32_16x16x32_bf16(false, ap, false, bv,
                                                       (short)0, acc[c], false, false);
    }
  }

  const int bb = bh >> 4, h = bh & 15;
#pragma unroll
  for (int v = 0; v < 8; v++) {
    float linv = 1.0f / fmaxf(lrow[v], 1e-12f);
    int t = q0 + v + 8 * half;
    size_t rowbase = (size_t)(bb * SEQ_T + t) * D_MODEL + h * D_HEAD;
#pragma unroll
    for (int c = 0; c < 4; c++)
      O[rowbase + c * 16 + l15] = f2bf(acc[c][v] * linv);
  }
}

// ---------------------------------------------------------------------------
// Launch: convert to bf16 -> Q/K/V projections -> flash attention -> Wo proj.
// Workspace (bf16): Xb 4M | Wq,Wk,Wv,Wo 1M each | Q,K,Vt,O 4M each = 48 MB.
// ---------------------------------------------------------------------------
extern "C" void kernel_launch(void* const* d_in, const int* in_sizes, int n_in,
                              void* d_out, int out_size, void* d_ws, size_t ws_size,
                              hipStream_t stream) {
  const float* x  = (const float*)d_in[0];
  const float* Wq = (const float*)d_in[1];
  const float* Wk = (const float*)d_in[2];
  const float* Wv = (const float*)d_in[3];
  const float* Wo = (const float*)d_in[4];
  float* out = (float*)d_out;

  const size_t EX = (size_t)M_ROWS * D_MODEL;    // 4M
  const size_t EW = (size_t)D_MODEL * D_MODEL;   // 1M
  bf16_t* Xb  = (bf16_t*)d_ws;
  bf16_t* Wqb = Xb  + EX;
  bf16_t* Wkb = Wqb + EW;
  bf16_t* Wvb = Wkb + EW;
  bf16_t* Wob = Wvb + EW;
  bf16_t* Qw  = Wob + EW;
  bf16_t* Kw  = Qw + EX;
  bf16_t* Vw  = Kw + EX;
  bf16_t* Ow  = Vw + EX;

  dim3 blk(256);
  cvt_bf16<<<(int)(EX / 1024), blk, 0, stream>>>(x,  Xb,  (int)EX);
  cvt_bf16<<<(int)(EW / 1024), blk, 0, stream>>>(Wq, Wqb, (int)EW);
  cvt_bf16<<<(int)(EW / 1024), blk, 0, stream>>>(Wk, Wkb, (int)EW);
  cvt_bf16<<<(int)(EW / 1024), blk, 0, stream>>>(Wv, Wvb, (int)EW);
  cvt_bf16<<<(int)(EW / 1024), blk, 0, stream>>>(Wo, Wob, (int)EW);

  dim3 gg(M_ROWS / 128, D_MODEL / 128);          // 32 x 8
  gemm_wmma<<<gg, blk, 0, stream>>>(Xb, Wqb, (void*)Qw, 1);
  gemm_wmma<<<gg, blk, 0, stream>>>(Xb, Wkb, (void*)Kw, 1);
  gemm_wmma<<<gg, blk, 0, stream>>>(Xb, Wvb, (void*)Vw, 2);

  dim3 ga(BATCH * N_HEADS, SEQ_T / (16 * 8));    // 32 x 16
  flash_attn<<<ga, blk, 0, stream>>>(Qw, Kw, Vw, Ow);

  gemm_wmma<<<gg, blk, 0, stream>>>(Ow, Wob, (void*)out, 0);
}
